// SimpleTextClassification_88579405512981
// MI455X (gfx1250) — matrix operand order repaired and checked by reading, not compiled, not run on hardware
//
#include <hip/hip_runtime.h>

// ---------------------------------------------------------------------------
// SimpleTextClassification on MI455X (gfx1250):
//   persistent fused LSTM, LDS-resident bf16 weights, v_wmma_f32_16x16x32_bf16,
//   async global->LDS h broadcast overlapped with the Wx GEMM half,
//   double-buffered x prefetch, one grid barrier per timestep,
//   WMMA MLP + softmax head.
// ---------------------------------------------------------------------------

#define B_      64
#define T_      512
#define E_      300
#define EP_     320      // E padded to multiple of 32 for K-chunks
#define U_      512
#define G_      2048     // 4*U
#define NWG_    32       // persistent workgroups; each owns 16 hidden units
#define UPW_    16       // units per workgroup
#define D1_     1024
#define D2_     1024
#define NC_     20

#if defined(__AMDGCN__) && __has_builtin(__builtin_amdgcn_global_load_async_to_lds_b128)
#define HAS_ASYNC_LDS 1
#else
#define HAS_ASYNC_LDS 0
#endif

typedef __attribute__((ext_vector_type(16))) __bf16 v16bf;
typedef __attribute__((ext_vector_type(8)))  float  v8f;
typedef __attribute__((ext_vector_type(4)))  int    v4i_;
#if HAS_ASYNC_LDS
typedef __attribute__((address_space(1))) v4i_* gp_v4i;   // global v4i*
typedef __attribute__((address_space(3))) v4i_* lp_v4i;   // LDS v4i*
#endif

union Frag { v16bf v; uint4 q[2]; };

__device__ __forceinline__ unsigned short f2bf(float f) {
    unsigned u = __float_as_uint(f);
    u += 0x7FFFu + ((u >> 16) & 1u);          // round-to-nearest-even
    return (unsigned short)(u >> 16);
}
__device__ __forceinline__ float bf2f(unsigned short s) {
    return __uint_as_float(((unsigned)s) << 16);
}

__device__ __forceinline__ v8f wmma_bf16(v16bf a, v16bf b, v8f c) {
    return __builtin_amdgcn_wmma_f32_16x16x32_bf16(false, a, false, b, (short)0, c,
                                                   false, false);
}

// A fragment (16x32 bf16), src row-major [rows][ldk], ldk multiple of 8.
// lane<16: row=lane, K = k0+0..7 / k0+16..23 ; lane>=16: K = k0+8..15 / k0+24..31
__device__ __forceinline__ v16bf load_a(const unsigned short* s, int ldk,
                                        int rt, int k0, int lane) {
    int row = rt * 16 + (lane & 15);
    int kh  = (lane >> 4) * 8;
    Frag f;
    f.q[0] = *(const uint4*)(s + row * ldk + k0 + kh);
    f.q[1] = *(const uint4*)(s + row * ldk + k0 + 16 + kh);
    return f.v;
}
// B fragment (32x16 bf16), src stored K-contiguous per column: wT[col][ldk]
__device__ __forceinline__ v16bf load_b(const unsigned short* s, int ldk,
                                        int ct, int k0, int lane) {
    int col = ct * 16 + (lane & 15);
    int kk  = k0 + (lane >> 4) * 8;
    Frag f;
    f.q[0] = *(const uint4*)(s + col * ldk + kk);
    f.q[1] = *(const uint4*)(s + col * ldk + kk + 16);
    return f.v;
}

// ---------------------------------------------------------------------------
// Persistent fused LSTM.  grid = NWG_ x 256 threads.
// LDS layout (bytes):
//   x_s0 [64][320] bf16  @ 0        (40960)   x_t double buffer
//   x_s1 [64][320] bf16  @ 40960    (40960)
//   h_s  [64][512] bf16  @ 81920    (65536)
//   wxT  [64][320] bf16  @ 147456   (40960)   Wx column slice, K-contiguous
//   whT  [64][512] bf16  @ 188416   (65536)   Wh column slice, K-contiguous
//   g_s  [64][64]  f32   @ 253952   (16384)
//   c_s  [64][16]  f32   @ 270336   (4096)
//   hl_s [64][16]  f32   @ 274432   (4096)
//   tk_s [2][64]   int   @ 278528   (512)
#define LSTM_LDS_BYTES 279040

__device__ __forceinline__ void stage_x(unsigned short* xb,
                                        const int* __restrict__ tokens,
                                        const float* __restrict__ emb,
                                        int t, int tid) {
    for (int idx = tid; idx < B_ * EP_; idx += 256) {
        int bb = idx / EP_, k = idx % EP_;
        float v = 0.f;
        if (k < E_) {
            int tok = tokens[bb * T_ + t];
            v = emb[tok * E_ + k];
        }
        xb[idx] = f2bf(v);
    }
}

__global__ __launch_bounds__(256, 1)
void lstm_persistent_kernel(const int* __restrict__ tokens,
                            const float* __restrict__ emb,
                            const float* __restrict__ Wx,
                            const float* __restrict__ Wh,
                            const float* __restrict__ bvec,
                            unsigned short* __restrict__ hg,   // 2 x [64][512] bf16
                            unsigned* __restrict__ cnt) {
    extern __shared__ __align__(16) char smem[];
    unsigned short* x_s0 = (unsigned short*)(smem);
    unsigned short* x_s1 = (unsigned short*)(smem + 40960);
    unsigned short* h_s  = (unsigned short*)(smem + 81920);
    unsigned short* wxT  = (unsigned short*)(smem + 147456);
    unsigned short* whT  = (unsigned short*)(smem + 188416);
    float*          g_s  = (float*)(smem + 253952);
    float*          c_s  = (float*)(smem + 270336);
    float*          hl_s = (float*)(smem + 274432);
    int*            tk_s = (int*)(smem + 278528);     // [2][64]

    const int tid  = threadIdx.x;
    const int wg   = blockIdx.x;
    const int lane = tid & 31;
    const int wv   = tid >> 5;
    const int ct   = wv & 3;            // gate block: 0=i 1=f 2=cc 3=o
    const int rt0  = (wv >> 2) * 2;     // this wave's two row tiles

    // ---- stage resident bf16 weight slices (one-time transpose gather) ----
    for (int idx = tid; idx < 64 * EP_; idx += 256) {
        int c = idx / EP_, k = idx % EP_;
        int gcol = (c >> 4) * U_ + wg * UPW_ + (c & 15);
        float w = (k < E_) ? Wx[k * G_ + gcol] : 0.f;
        wxT[c * EP_ + k] = f2bf(w);
    }
    for (int idx = tid; idx < 64 * U_; idx += 256) {
        int c = idx / U_, k = idx % U_;
        int gcol = (c >> 4) * U_ + wg * UPW_ + (c & 15);
        whT[c * U_ + k] = f2bf(Wh[k * G_ + gcol]);
    }
    for (int idx = tid; idx < B_ * UPW_; idx += 256) {
        c_s[idx] = 0.f;
        hl_s[idx] = 0.f;
    }
    // prologue: x_0 and tokens_0
    stage_x(x_s0, tokens, emb, 0, tid);
    if (tid < B_) tk_s[tid] = tokens[tid * T_ + 0];
    const float bias = bvec[ct * U_ + wg * UPW_ + (lane & 15)];
    __syncthreads();

    for (int t = 0; t < T_; ++t) {
        const unsigned short* xc  = (t & 1) ? x_s1 : x_s0;
        unsigned short*       xn  = (t & 1) ? x_s0 : x_s1;
        const int*            tkc = tk_s + (t & 1) * 64;
        int*                  tkn = tk_s + ((t + 1) & 1) * 64;
        const unsigned short* hrd = hg + (t & 1) * (B_ * U_);
        unsigned short*       hwr = hg + ((t + 1) & 1) * (B_ * U_);

        // ---- (1) launch the 64KB h broadcast (global -> LDS) ----
#if HAS_ASYNC_LDS
        {
            const char* src = (const char*)hrd;
            char*       dst = (char*)h_s;
            for (int i = tid; i < (B_ * U_) / 8; i += 256) {
                __builtin_amdgcn_global_load_async_to_lds_b128(
                    (gp_v4i)(src + 16 * i),
                    (lp_v4i)(dst + 16 * i),
                    0, 0);
            }
        }
#else
        {
            const uint4* src = (const uint4*)hrd;
            uint4*       dst = (uint4*)h_s;
            for (int i = tid; i < (B_ * U_) / 8; i += 256) dst[i] = src[i];
        }
#endif

        // ---- (2) Wx half of the GEMM while the h copy is in flight ----
        v8f acc0, acc1;
        for (int v = 0; v < 8; ++v) { acc0[v] = bias; acc1[v] = bias; }
        for (int k0 = 0; k0 < EP_; k0 += 32) {
            v16bf bf = load_b(wxT, EP_, ct, k0, lane);
            v16bf a0 = load_a(xc, EP_, rt0, k0, lane);
            v16bf a1 = load_a(xc, EP_, rt0 + 1, k0, lane);
            acc0 = wmma_bf16(a0, bf, acc0);
            acc1 = wmma_bf16(a1, bf, acc1);
        }

        // ---- h copy completion + block barrier ----
#if HAS_ASYNC_LDS
#if __has_builtin(__builtin_amdgcn_s_wait_asynccnt)
        __builtin_amdgcn_s_wait_asynccnt(0);
#else
        asm volatile("s_wait_asynccnt 0x0" ::: "memory");
#endif
#endif
        __syncthreads();

        // ---- (3) Wh half of the GEMM ----
        for (int k0 = 0; k0 < U_; k0 += 32) {
            v16bf bf = load_b(whT, U_, ct, k0, lane);
            v16bf a0 = load_a(h_s, U_, rt0, k0, lane);
            v16bf a1 = load_a(h_s, U_, rt0 + 1, k0, lane);
            acc0 = wmma_bf16(a0, bf, acc0);
            acc1 = wmma_bf16(a1, bf, acc1);
        }
        {
            int nn = lane & 15, mb = (lane >> 4) * 8;
            for (int v = 0; v < 8; ++v) {
                g_s[(rt0 * 16 + mb + v) * 64 + ct * 16 + nn]       = acc0[v];
                g_s[((rt0 + 1) * 16 + mb + v) * 64 + ct * 16 + nn] = acc1[v];
            }
        }
        __syncthreads();

        // ---- (4) gates + masked state update (f32) ----
        for (int idx = tid; idx < B_ * UPW_; idx += 256) {
            int bb = idx >> 4, u = idx & 15;
            float gi = g_s[bb * 64 + u];
            float gf = g_s[bb * 64 + 16 + u];
            float gc = g_s[bb * 64 + 32 + u];
            float go = g_s[bb * 64 + 48 + u];
            float si = 1.f / (1.f + __expf(-gi));
            float sf = 1.f / (1.f + __expf(-gf));
            float so = 1.f / (1.f + __expf(-go));
            float cc = fmaxf(gc, 0.f);
            float cold = c_s[idx];
            float cnew = sf * cold + si * cc;
            float hnew = so * fmaxf(cnew, 0.f);
            bool  m  = (tkc[bb] != 0);
            float hu = m ? hnew : hl_s[idx];
            float cu = m ? cnew : cold;
            c_s[idx]  = cu;
            hl_s[idx] = hu;
            hwr[bb * U_ + wg * UPW_ + u] = f2bf(hu);
        }

        // ---- (5) prefetch x_{t+1} + tokens_{t+1} off the critical path ----
        if (t + 1 < T_) {
            stage_x(xn, tokens, emb, t + 1, tid);
            if (tid < B_) tkn[tid] = tokens[tid * T_ + (t + 1)];
        }

        // ---- (6) grid-wide step barrier (agent-scope atomic counter) ----
        __threadfence();
        __syncthreads();
        if (tid == 0) {
            __hip_atomic_fetch_add(cnt, 1u, __ATOMIC_RELEASE, __HIP_MEMORY_SCOPE_AGENT);
            unsigned target = (unsigned)NWG_ * (unsigned)(t + 1);
            while (__hip_atomic_load(cnt, __ATOMIC_ACQUIRE, __HIP_MEMORY_SCOPE_AGENT) < target)
                __builtin_amdgcn_s_sleep(2);
        }
        __syncthreads();
    }
    // final h (t=511 writes buffer (512)&1 = 0) lives at hg[0..B*U)
}

// ---------------------------------------------------------------------------
// Dense layer: Out[64][N] = relu(A[64][K](bf16) @ W[K][N](f32) + bias), bf16 out.
// grid.x = N/64 blocks; each block = 64x64 tile, 8 waves x 2 WMMA tiles.
__global__ __launch_bounds__(256, 1)
void fc_relu_kernel(const unsigned short* __restrict__ A,
                    const float* __restrict__ W,
                    const float* __restrict__ bias,
                    unsigned short* __restrict__ Out,
                    int K, int N) {
    extern __shared__ __align__(16) char smem[];
    unsigned short* a_s = (unsigned short*)smem;
    unsigned short* wT  = (unsigned short*)(smem + (size_t)64 * K * 2);

    const int tid = threadIdx.x;
    const int blk = blockIdx.x;

    {   // stage A (vectorized, already bf16)
        const uint4* src = (const uint4*)A;
        uint4*       dst = (uint4*)a_s;
        int n4 = (64 * K) / 8;
        for (int i = tid; i < n4; i += 256) dst[i] = src[i];
    }
    for (int idx = tid; idx < 64 * K; idx += 256) {   // transpose + cvt weights
        int c = idx / K, k = idx % K;
        wT[idx] = f2bf(W[k * N + blk * 64 + c]);
    }
    __syncthreads();

    const int lane = tid & 31, wv = tid >> 5;
    const int ct = wv & 3, rt0 = (wv >> 2) * 2;

    v8f acc0 = {0.f, 0.f, 0.f, 0.f, 0.f, 0.f, 0.f, 0.f};
    v8f acc1 = {0.f, 0.f, 0.f, 0.f, 0.f, 0.f, 0.f, 0.f};
    for (int k0 = 0; k0 < K; k0 += 32) {
        v16bf bf = load_b(wT, K, ct, k0, lane);
        v16bf a0 = load_a(a_s, K, rt0, k0, lane);
        v16bf a1 = load_a(a_s, K, rt0 + 1, k0, lane);
        acc0 = wmma_bf16(a0, bf, acc0);
        acc1 = wmma_bf16(a1, bf, acc1);
    }
    const int col = blk * 64 + ct * 16 + (lane & 15);
    const float bi = bias[col];
    const int mb = (lane >> 4) * 8;
    for (int v = 0; v < 8; ++v) {
        int r0 = rt0 * 16 + mb + v;
        int r1 = r0 + 16;
        Out[r0 * N + col] = f2bf(fmaxf(acc0[v] + bi, 0.f));
        Out[r1 * N + col] = f2bf(fmaxf(acc1[v] + bi, 0.f));
    }
}

// ---------------------------------------------------------------------------
// Head: logits = H[64][1024](bf16) @ Wo[1024][20] + bo ; per-row softmax -> f32.
__global__ __launch_bounds__(256, 1)
void head_softmax_kernel(const unsigned short* __restrict__ H,
                         const float* __restrict__ Wo,
                         const float* __restrict__ bo,
                         float* __restrict__ out) {
    __shared__ float lg[16 * NC_];
    const int tid = threadIdx.x;
    const int rb = blockIdx.x * 16;

    for (int idx = tid; idx < 16 * NC_; idx += 256) {
        int r = idx / NC_, cl = idx % NC_;
        int b = rb + r;
        float s = bo[cl];
        for (int k = 0; k < D2_; ++k)
            s += bf2f(H[b * D2_ + k]) * Wo[k * NC_ + cl];
        lg[idx] = s;
    }
    __syncthreads();
    if (tid < 16) {
        float mx = -1e30f;
        for (int cl = 0; cl < NC_; ++cl) mx = fmaxf(mx, lg[tid * NC_ + cl]);
        float e[NC_];
        float sm = 0.f;
        for (int cl = 0; cl < NC_; ++cl) {
            e[cl] = __expf(lg[tid * NC_ + cl] - mx);
            sm += e[cl];
        }
        float inv = 1.f / sm;
        for (int cl = 0; cl < NC_; ++cl) out[(rb + tid) * NC_ + cl] = e[cl] * inv;
    }
}

// ---------------------------------------------------------------------------
extern "C" void kernel_launch(void* const* d_in, const int* in_sizes, int n_in,
                              void* d_out, int out_size, void* d_ws, size_t ws_size,
                              hipStream_t stream) {
    const int*   tokens = (const int*)d_in[0];
    const float* emb    = (const float*)d_in[1];
    const float* Wx     = (const float*)d_in[2];
    const float* Wh     = (const float*)d_in[3];
    const float* bvec   = (const float*)d_in[4];
    const float* W1     = (const float*)d_in[5];
    const float* b1     = (const float*)d_in[6];
    const float* W2     = (const float*)d_in[7];
    const float* b2     = (const float*)d_in[8];
    const float* Wo     = (const float*)d_in[9];
    const float* bo     = (const float*)d_in[10];
    float* out = (float*)d_out;

    char* ws = (char*)d_ws;
    unsigned short* hg  = (unsigned short*)ws;                    // 2*64*512 bf16 = 131072 B
    unsigned*       cnt = (unsigned*)(ws + 131072);               // counter (pad to 128)
    unsigned short* o1  = (unsigned short*)(ws + 131200);         // 64*1024 bf16
    unsigned short* o2  = (unsigned short*)(ws + 131200 + 131072);// 64*1024 bf16

    // zero h double-buffer + barrier counter every launch (capture-safe)
    (void)hipMemsetAsync(ws, 0, 131200, stream);

    lstm_persistent_kernel<<<NWG_, 256, LSTM_LDS_BYTES, stream>>>(
        tokens, emb, Wx, Wh, bvec, hg, cnt);

    fc_relu_kernel<<<D1_ / 64, 256, 2 * 64 * U_ * 2, stream>>>(hg, W1, b1, o1, U_, D1_);
    fc_relu_kernel<<<D2_ / 64, 256, 2 * 64 * D1_ * 2, stream>>>(o1, W2, b2, o2, D1_, D2_);
    head_softmax_kernel<<<B_ / 16, 256, 0, stream>>>(o2, Wo, bo, out);
}
